// leaky_integrator_VAE_68032281969205
// MI455X (gfx1250) — matrix operand order repaired and checked by reading, not compiled, not run on hardware
//
#include <hip/hip_runtime.h>
#include <stdint.h>

typedef __attribute__((ext_vector_type(2)))  float   v2f;
typedef __attribute__((ext_vector_type(8)))  float   v8f;
typedef __attribute__((ext_vector_type(16))) __bf16  v16bf;

#define NN 1024
#define TT 16384
#define DT_C 0.01f

// ---------------- bf16 split helpers (round-to-nearest-even) ----------------
__device__ __forceinline__ uint32_t bf16r(float v) {
  uint32_t u = __float_as_uint(v);
  return (u + 0x7fffu + ((u >> 16) & 1u)) >> 16;
}
__device__ __forceinline__ float bf2f(uint32_t h) { return __uint_as_float(h << 16); }

// ---------------------------------------------------------------------------
// Prep: W_c = sparsity_c * max(mag_c,0) * signs_c
//       W_e'[i,j] = sparsity_e[i,j]*(max(me[i,j],0)+max(me[j,i],0)) - delta_ij*rowsum_i
//       dtt[i] = DT / tau[i]
// ---------------------------------------------------------------------------
__global__ __launch_bounds__(256)
void prep_kernel(const float* __restrict__ sc, const float* __restrict__ se,
                 const float* __restrict__ sg, const float* __restrict__ mc,
                 const float* __restrict__ me, const float* __restrict__ tau,
                 float* __restrict__ wc, float* __restrict__ we,
                 float* __restrict__ dtt) {
  __shared__ float red[256];
  const int i = blockIdx.x;
  const int tid = threadIdx.x;
  float wloc[4];
  float sum = 0.f;
#pragma unroll
  for (int p = 0; p < 4; ++p) {
    const int j = tid + p * 256;
    const int ij = i * NN + j;
    const float m_c = fmaxf(mc[ij], 0.f);
    wc[ij] = sc[ij] * m_c * sg[ij];
    const float m_ij = fmaxf(me[ij], 0.f);
    const float m_ji = fmaxf(me[j * NN + i], 0.f);
    const float w = se[ij] * (m_ij + m_ji);
    wloc[p] = w;
    sum += w;
  }
  red[tid] = sum;
  __syncthreads();
  for (int s = 128; s > 0; s >>= 1) {
    if (tid < s) red[tid] += red[tid + s];
    __syncthreads();
  }
  const float rowsum = red[0];
#pragma unroll
  for (int p = 0; p < 4; ++p) {
    const int j = tid + p * 256;
    we[i * NN + j] = wloc[p] - ((j == i) ? rowsum : 0.f);
  }
  if (tid == 0) dtt[i] = DT_C / tau[i];
}

// ---------------------------------------------------------------------------
// Split W into bf16 hi/lo, pre-swizzled into this code's contiguous-k A-fragment order (k = 16*half + s):
// per (row i, K-block of 32): 32 bf16 stored as [half0: K{0..15}][half1: K{16..31}]
// so each lane's v16bf fragment is one contiguous 32B load.
// One thread per K-pair (k even): handles W_c and W_e'.
// ---------------------------------------------------------------------------
__global__ __launch_bounds__(256)
void split_w_kernel(const float* __restrict__ wcf, const float* __restrict__ wef,
                    uint32_t* __restrict__ wch, uint32_t* __restrict__ wcl,
                    uint32_t* __restrict__ weh, uint32_t* __restrict__ wel) {
  const int gid = blockIdx.x * 256 + threadIdx.x;   // pair id: N*N/2 total
  const int i  = gid >> 9;                          // 512 pairs per row
  const int k  = (gid & 511) * 2;
  const int blk = i * 32 + (k >> 5);
  const int h   = (k >> 4) & 1;
  const int p   = k & 15;                           // even since k even
  const size_t dst = (size_t)blk * 16 + h * 8 + (p >> 1);   // u32 units
  const size_t src = (size_t)i * NN + k;

  const float c0 = wcf[src], c1 = wcf[src + 1];
  const float e0 = wef[src], e1 = wef[src + 1];
  const uint32_t ch0 = bf16r(c0), ch1 = bf16r(c1);
  const uint32_t eh0 = bf16r(e0), eh1 = bf16r(e1);
  wch[dst] = ch0 | (ch1 << 16);
  weh[dst] = eh0 | (eh1 << 16);
  const uint32_t cl0 = bf16r(c0 - bf2f(ch0)), cl1 = bf16r(c1 - bf2f(ch1));
  const uint32_t el0 = bf16r(e0 - bf2f(eh0)), el1 = bf16r(e1 - bf2f(eh1));
  wcl[dst] = cl0 | (cl1 << 16);
  wel[dst] = el0 | (el1 << 16);
}

// ---------------------------------------------------------------------------
// Build transposed bf16 splits of x (and relu(x)): xT[t][k] = x[k][t],
// x[k][t] = (t==0) ? 0 : hidden[k*T + t-1].  Arrays [T][N] bf16, packed b32.
// Block tile: 32 k x 64 t, LDS transpose for coalesced loads and stores.
// ---------------------------------------------------------------------------
__global__ __launch_bounds__(256)
void split_x_kernel(const float* __restrict__ hidden,
                    uint32_t* __restrict__ xth, uint32_t* __restrict__ xtl,
                    uint32_t* __restrict__ rth, uint32_t* __restrict__ rtl) {
  __shared__ float tile[64][34];
  const int tid = threadIdx.x;
  const int t0  = blockIdx.x * 64;
  const int k0  = blockIdx.y * 32;

  const int tl = tid & 63, kq = tid >> 6;           // load: coalesced along t
  const int tg = t0 + tl;
#pragma unroll
  for (int p = 0; p < 8; ++p) {
    const int k = kq * 8 + p;
    float v = 0.f;
    if (tg > 0) v = hidden[(size_t)(k0 + k) * TT + tg - 1];
    tile[tl][k] = v;
  }
  __syncthreads();

  const int k2 = tid & 15, tq = tid >> 4;           // store: coalesced along k
#pragma unroll
  for (int q = 0; q < 4; ++q) {
    const int t = tq + 16 * q;
    const float v0 = tile[t][2 * k2];
    const float v1 = tile[t][2 * k2 + 1];
    const size_t idx = ((size_t)(t0 + t) * NN + k0 + 2 * k2) >> 1;  // u32 units
    const uint32_t h0 = bf16r(v0), h1 = bf16r(v1);
    xth[idx] = h0 | (h1 << 16);
    const uint32_t l0 = bf16r(v0 - bf2f(h0)), l1 = bf16r(v1 - bf2f(h1));
    xtl[idx] = l0 | (l1 << 16);
    const float r0 = fmaxf(v0, 0.f), r1 = fmaxf(v1, 0.f);
    const uint32_t q0 = bf16r(r0), q1 = bf16r(r1);
    rth[idx] = q0 | (q1 << 16);
    const uint32_t s0 = bf16r(r0 - bf2f(q0)), s1 = bf16r(r1 - bf2f(q1));
    rtl[idx] = s0 | (s1 << 16);
  }
}

// ---------------------------------------------------------------------------
// Split-precision bf16 GEMM with V_WMMA_F32_16X16X32_BF16:
// recurrent = (Wc_h+Wc_l)@(r_h+r_l) + (We_h+We_l)@(x_h+x_l), 3 products each
// (hi*hi + hi*lo + lo*hi), f32 accumulate -> ~f32-GEMM accuracy.
// Block: 8 waves; each wave: 2 M-tiles x 2 T-tiles = 4 accumulators.
// Block tile: 32 rows x 256 time. Grid: (T/256, N/32) = (64, 32).
// A fragment: one 32B contiguous load (pre-swizzled). B fragment: one 32B
// contiguous load from transposed arrays (lane t = l16, K-half by lane>>4).
// ---------------------------------------------------------------------------
#define WMMA_BF16(A, B, C) __builtin_amdgcn_wmma_f32_16x16x32_bf16( \
    false, (A), false, (B), (short)0, (C), false, false)

__global__ __launch_bounds__(256)
void gemm_bf16_kernel(const float* __restrict__ input, const float* __restrict__ hidden,
                      const float* __restrict__ bias, const float* __restrict__ dtt,
                      const __bf16* __restrict__ wch, const __bf16* __restrict__ wcl,
                      const __bf16* __restrict__ weh, const __bf16* __restrict__ wel,
                      const __bf16* __restrict__ xth, const __bf16* __restrict__ xtl,
                      const __bf16* __restrict__ rth, const __bf16* __restrict__ rtl,
                      float* __restrict__ out) {
  const int tid  = threadIdx.x;
  const int lane = tid & 31;
  const int wave = tid >> 5;    // 0..7 : T strip
  const int half = lane >> 4;
  const int l16  = lane & 15;

  const int i0 = blockIdx.y * 32;
  const int t0 = blockIdx.x * 256 + wave * 32;

  // A: swizzled offset = row*1024 + half*16 + kb32*32   (bf16 units)
  const int row0 = i0 + l16;
  size_t a0 = (size_t)row0 * 1024 + half * 16;
  size_t a1 = a0 + (size_t)16 * 1024;
  // B: transposed offset = t*1024 + half*16 + kb32*32
  size_t b0 = (size_t)(t0 + l16) * 1024 + half * 16;
  size_t b1 = b0 + (size_t)16 * 1024;

  v8f acc[2][2];
#pragma unroll
  for (int m = 0; m < 2; ++m)
#pragma unroll
    for (int s = 0; s < 2; ++s) acc[m][s] = 0.f;

  for (int kb = 0; kb < NN / 32; ++kb, a0 += 32, a1 += 32, b0 += 32, b1 += 32) {
    // ---- chem: Wc x relu(x) ----
    const v16bf Bh0 = *(const v16bf*)(rth + b0);
    const v16bf Bh1 = *(const v16bf*)(rth + b1);
    const v16bf Ah0 = *(const v16bf*)(wch + a0);
    const v16bf Ah1 = *(const v16bf*)(wch + a1);
    acc[0][0] = WMMA_BF16(Ah0, Bh0, acc[0][0]);
    acc[0][1] = WMMA_BF16(Ah0, Bh1, acc[0][1]);
    acc[1][0] = WMMA_BF16(Ah1, Bh0, acc[1][0]);
    acc[1][1] = WMMA_BF16(Ah1, Bh1, acc[1][1]);
    const v16bf Bl0 = *(const v16bf*)(rtl + b0);
    const v16bf Bl1 = *(const v16bf*)(rtl + b1);
    acc[0][0] = WMMA_BF16(Ah0, Bl0, acc[0][0]);
    acc[0][1] = WMMA_BF16(Ah0, Bl1, acc[0][1]);
    acc[1][0] = WMMA_BF16(Ah1, Bl0, acc[1][0]);
    acc[1][1] = WMMA_BF16(Ah1, Bl1, acc[1][1]);
    const v16bf Al0 = *(const v16bf*)(wcl + a0);
    const v16bf Al1 = *(const v16bf*)(wcl + a1);
    acc[0][0] = WMMA_BF16(Al0, Bh0, acc[0][0]);
    acc[0][1] = WMMA_BF16(Al0, Bh1, acc[0][1]);
    acc[1][0] = WMMA_BF16(Al1, Bh0, acc[1][0]);
    acc[1][1] = WMMA_BF16(Al1, Bh1, acc[1][1]);
    // ---- elec: We' x x ----
    const v16bf Eh0 = *(const v16bf*)(xth + b0);
    const v16bf Eh1 = *(const v16bf*)(xth + b1);
    const v16bf Fh0 = *(const v16bf*)(weh + a0);
    const v16bf Fh1 = *(const v16bf*)(weh + a1);
    acc[0][0] = WMMA_BF16(Fh0, Eh0, acc[0][0]);
    acc[0][1] = WMMA_BF16(Fh0, Eh1, acc[0][1]);
    acc[1][0] = WMMA_BF16(Fh1, Eh0, acc[1][0]);
    acc[1][1] = WMMA_BF16(Fh1, Eh1, acc[1][1]);
    const v16bf El0 = *(const v16bf*)(xtl + b0);
    const v16bf El1 = *(const v16bf*)(xtl + b1);
    acc[0][0] = WMMA_BF16(Fh0, El0, acc[0][0]);
    acc[0][1] = WMMA_BF16(Fh0, El1, acc[0][1]);
    acc[1][0] = WMMA_BF16(Fh1, El0, acc[1][0]);
    acc[1][1] = WMMA_BF16(Fh1, El1, acc[1][1]);
    const v16bf Fl0 = *(const v16bf*)(wel + a0);
    const v16bf Fl1 = *(const v16bf*)(wel + a1);
    acc[0][0] = WMMA_BF16(Fl0, Eh0, acc[0][0]);
    acc[0][1] = WMMA_BF16(Fl0, Eh1, acc[0][1]);
    acc[1][0] = WMMA_BF16(Fl1, Eh0, acc[1][0]);
    acc[1][1] = WMMA_BF16(Fl1, Eh1, acc[1][1]);
  }

  // Epilogue (exact f32 x / input): C/D layout row = i0+m*16+half*8+r, col t.
  const size_t NT = (size_t)NN * TT;
#pragma unroll
  for (int m = 0; m < 2; ++m) {
#pragma unroll
    for (int r = 0; r < 8; ++r) {
      const int i = i0 + m * 16 + half * 8 + r;
      const float dt = dtt[i];
      const float bs = bias[i];
      const size_t base = (size_t)i * TT;
#pragma unroll
      for (int s = 0; s < 2; ++s) {
        const int t = t0 + s * 16 + l16;
        const float rec = acc[m][s][r];
        const float xv  = (t == 0) ? 0.f : hidden[base + t - 1];
        const float mu  = dt * (rec + bs + input[base + t] - xv) + xv;
        out[base + t]      = mu;
        out[NT + base + t] = rec;
      }
    }
  }
}

// ---------------------------------------------------------------------------
// Fallback f32 WMMA GEMM (used when workspace is too small for bf16 staging).
// ---------------------------------------------------------------------------
__global__ __launch_bounds__(256)
void gemm_f32_kernel(const float* __restrict__ input, const float* __restrict__ hidden,
                     const float* __restrict__ bias,
                     const float* __restrict__ wc, const float* __restrict__ we,
                     const float* __restrict__ dtt, float* __restrict__ out) {
  const int tid  = threadIdx.x;
  const int lane = tid & 31;
  const int wave = tid >> 5;
  const int half = lane >> 4;
  const int l16  = lane & 15;

  const int i0 = blockIdx.y * 32;
  const int t0 = blockIdx.x * 512 + wave * 64;

  const int arow = i0 + l16;
  const float* __restrict__ wc0 = wc + (size_t)arow * NN + half * 2;
  const float* __restrict__ we0 = we + (size_t)arow * NN + half * 2;
  const float* __restrict__ wc1 = wc0 + (size_t)16 * NN;
  const float* __restrict__ we1 = we0 + (size_t)16 * NN;

  float msk[4];
  int   tIdx[4];
#pragma unroll
  for (int s = 0; s < 4; ++s) {
    const int t = t0 + s * 16 + l16;
    msk[s]  = (t == 0) ? 0.f : 1.f;
    tIdx[s] = (t == 0) ? 0 : (t - 1);
  }

  v8f acc[2][4];
#pragma unroll
  for (int m = 0; m < 2; ++m)
#pragma unroll
    for (int s = 0; s < 4; ++s) acc[m][s] = 0.f;

  const int kB = half * 2;
  for (int kb = 0; kb < NN; kb += 4) {
    const v2f ac0 = *(const v2f*)(wc0 + kb);
    const v2f ae0 = *(const v2f*)(we0 + kb);
    const v2f ac1 = *(const v2f*)(wc1 + kb);
    const v2f ae1 = *(const v2f*)(we1 + kb);
    const float* __restrict__ xrow0 = hidden + (size_t)(kb + kB) * TT;
    const float* __restrict__ xrow1 = xrow0 + TT;
#pragma unroll
    for (int s = 0; s < 4; ++s) {
      const float b0 = msk[s] * xrow0[tIdx[s]];
      const float b1 = msk[s] * xrow1[tIdx[s]];
      const v2f be = {b0, b1};
      const v2f br = {fmaxf(b0, 0.f), fmaxf(b1, 0.f)};
      acc[0][s] = __builtin_amdgcn_wmma_f32_16x16x4_f32(false, ac0, false, br, (short)0, acc[0][s], false, false);
      acc[0][s] = __builtin_amdgcn_wmma_f32_16x16x4_f32(false, ae0, false, be, (short)0, acc[0][s], false, false);
      acc[1][s] = __builtin_amdgcn_wmma_f32_16x16x4_f32(false, ac1, false, br, (short)0, acc[1][s], false, false);
      acc[1][s] = __builtin_amdgcn_wmma_f32_16x16x4_f32(false, ae1, false, be, (short)0, acc[1][s], false, false);
    }
  }

  const size_t NT = (size_t)NN * TT;
#pragma unroll
  for (int m = 0; m < 2; ++m) {
#pragma unroll
    for (int r = 0; r < 8; ++r) {
      const int i = i0 + m * 16 + half * 8 + r;
      const float dt = dtt[i];
      const float bs = bias[i];
      const size_t base = (size_t)i * TT;
#pragma unroll
      for (int s = 0; s < 4; ++s) {
        const int t = t0 + s * 16 + l16;
        const float rec = acc[m][s][r];
        const float xv  = msk[s] * hidden[base + tIdx[s]];
        const float mu  = dt * (rec + bs + input[base + t] - xv) + xv;
        out[base + t]      = mu;
        out[NT + base + t] = rec;
      }
    }
  }
}

// ---------------------------------------------------------------------------
extern "C" void kernel_launch(void* const* d_in, const int* in_sizes, int n_in,
                              void* d_out, int out_size, void* d_ws, size_t ws_size,
                              hipStream_t stream) {
  const float* input  = (const float*)d_in[0];
  const float* hidden = (const float*)d_in[1];
  const float* sc     = (const float*)d_in[2];
  const float* se     = (const float*)d_in[3];
  const float* sg     = (const float*)d_in[4];
  const float* mc     = (const float*)d_in[5];
  const float* me     = (const float*)d_in[6];
  const float* bias   = (const float*)d_in[7];
  const float* tau    = (const float*)d_in[8];
  float* out = (float*)d_out;

  // ws layout: [0,4M) Wc f32 | [4M,8M) We' f32 | [8M,+4K) dtt |
  //            [16M,24M) swizzled bf16 W (4 x 2MB) | [24M,152M) xT bf16 (4 x 32MB)
  float* wcf = (float*)d_ws;
  float* wef = wcf + (size_t)NN * NN;
  float* dtt = wef + (size_t)NN * NN;

  const size_t NTsz = (size_t)NN * TT;
  const size_t FB   = (size_t)16 << 20;
  const size_t NEED = FB + 4 * (size_t)NN * NN * 2 + 4 * NTsz * 2;  // 152 MB

  prep_kernel<<<NN, 256, 0, stream>>>(sc, se, sg, mc, me, tau, wcf, wef, dtt);

  if (ws_size >= NEED) {
    __bf16* wch = (__bf16*)((char*)d_ws + FB);
    __bf16* wcl = wch + (size_t)NN * NN;
    __bf16* weh = wcl + (size_t)NN * NN;
    __bf16* wel = weh + (size_t)NN * NN;
    __bf16* xth = wel + (size_t)NN * NN;
    __bf16* xtl = xth + NTsz;
    __bf16* rth = xtl + NTsz;
    __bf16* rtl = rth + NTsz;

    split_w_kernel<<<(NN * NN / 2) / 256, 256, 0, stream>>>(
        wcf, wef, (uint32_t*)wch, (uint32_t*)wcl, (uint32_t*)weh, (uint32_t*)wel);
    dim3 gx(TT / 64, NN / 32);
    split_x_kernel<<<gx, 256, 0, stream>>>(
        hidden, (uint32_t*)xth, (uint32_t*)xtl, (uint32_t*)rth, (uint32_t*)rtl);

    dim3 grid(TT / 256, NN / 32);
    gemm_bf16_kernel<<<grid, 256, 0, stream>>>(input, hidden, bias, dtt,
                                               wch, wcl, weh, wel,
                                               xth, xtl, rth, rtl, out);
  } else {
    dim3 grid(TT / 512, NN / 32);
    gemm_f32_kernel<<<grid, 256, 0, stream>>>(input, hidden, bias, wcf, wef, dtt, out);
  }
}
